// KoopmanOperator_68547678044258
// MI455X (gfx1250) — matrix operand order repaired and checked by reading, not compiled, not run on hardware
//
#include <hip/hip_runtime.h>

// ---------------------------------------------------------------------------
// Koopman operator forward, fused, CDNA5 (gfx1250) wave32 + v_wmma_f32_16x16x32_f16
// Two kernels: prep (swizzle weights -> f16 B-fragments in global ws, fold
// normalization into W1/B and bias vectors), then fused main kernel.
// ---------------------------------------------------------------------------
typedef _Float16 v16h __attribute__((ext_vector_type(16)));
typedef float    v8f  __attribute__((ext_vector_type(8)));

static constexpr int SDIM  = 32;
static constexpr int ADIM  = 8;
static constexpr int LDIM  = 64;
static constexpr int BATCH = 256;
static constexpr int SEQ   = 1024;
static constexpr int ROWS  = BATCH * SEQ;        // 262144
static constexpr int TILES = ROWS / 16;          // 16384
static constexpr int WPB   = 8;                  // waves per block
static constexpr int PITCH = 72;                 // f16 staging pitch (bank spread, 16B aligned)

// swizzled B-operand weight fragments (each: 32 lanes x 16 f16 = 512 halves)
static constexpr int F_W1 = 0;    // 1 Kchunk x 4 Ntiles  (normalization-folded)
static constexpr int F_W2 = 4;    // 2 x 4
static constexpr int F_W3 = 12;   // 2 x 4
static constexpr int F_A  = 20;   // 2 x 4
static constexpr int F_B  = 28;   // 1 x 4 (K=8 zero-padded to 32, normalization-folded)
static constexpr int F_C  = 32;   // 2 x 2
static constexpr int NFRAG = 36;

// d_ws layout (bytes)
static constexpr size_t WS_WF    = 0;                       // 36864 B of f16 fragments
static constexpr size_t WS_BIASS = (size_t)NFRAG * 512 * 2; // 64 f32 folded encoder bias
static constexpr size_t WS_BIASA = WS_BIASS + 64 * 4;       // 64 f32 folded action bias

__device__ __forceinline__ v8f wmma_f16(v16h a, v16h b, v8f c) {
  return __builtin_amdgcn_wmma_f32_16x16x32_f16(false, a, false, b, (short)0, c,
                                                false, false);
}

// B-operand fragment: lane-major contiguous in LDS
__device__ __forceinline__ v16h bfrag(const _Float16* wf, int f, int lane) {
  const _Float16* p = wf + f * 512 + lane * 16;
  v16h b;
#pragma unroll
  for (int i = 0; i < 16; ++i) b[i] = p[i];
  return b;
}

// A-operand fragment from row-major f16 staging tile.
// ISA 16-bit A 16x32 layout: lane kb = (lane>>4)*8; e<8 -> K=kb+e; e>=8 -> K=kb+16+(e-8)
__device__ __forceinline__ v16h afrag(const _Float16* st, int m, int kb, int kc) {
  const _Float16* p = st + m * PITCH + kc * 32 + kb;
  v16h a;
#pragma unroll
  for (int i = 0; i < 8; ++i) { a[i] = p[i]; a[i + 8] = p[16 + i]; }
  return a;
}

// ---------------------------------------------------------------------------
// Prep: build swizzled f16 B-fragments (normalization folded) + bias vectors.
// One block; weights are tiny (68 KB f32) and stay L2-resident.
// B-matrix VGPR layout (wave32): lane n = lane&15; K = (lane>>4)*8 + (e<8 ? e : e+8)
// ---------------------------------------------------------------------------
__global__ __launch_bounds__(256) void koopman_prep_kernel(
    const float* __restrict__ W1, const float* __restrict__ b1,
    const float* __restrict__ W2, const float* __restrict__ W3,
    const float* __restrict__ Aw, const float* __restrict__ Bw,
    const float* __restrict__ Cw,
    const float* __restrict__ s_shift, const float* __restrict__ s_scale,
    const float* __restrict__ a_shift, const float* __restrict__ a_scale,
    _Float16* __restrict__ wfg, float* __restrict__ biasS,
    float* __restrict__ biasA) {
  const int tid = threadIdx.x;
  for (int idx = tid; idx < NFRAG * 512; idx += 256) {
    const int f = idx >> 9, r = idx & 511, ln = r >> 4, e = r & 15;
    const int n = ln & 15;
    const int kk = ((ln >> 4) << 3) + (e < 8 ? e : e + 8); // K within 32-wide chunk
    float v;
    if (f < F_W2) {
      v = W1[kk * LDIM + f * 16 + n] / s_scale[kk];    // fold 1/state_scale
    } else if (f < F_W3) {
      const int g = f - F_W2;
      v = W2[((g >> 2) * 32 + kk) * LDIM + (g & 3) * 16 + n];
    } else if (f < F_A) {
      const int g = f - F_W3;
      v = W3[((g >> 2) * 32 + kk) * LDIM + (g & 3) * 16 + n];
    } else if (f < F_B) {
      const int g = f - F_A;
      v = Aw[((g >> 2) * 32 + kk) * LDIM + (g & 3) * 16 + n];
    } else if (f < F_C) {
      const int g = f - F_B;                            // fold 1/action_scale, pad K>=8
      v = (kk < ADIM) ? Bw[kk * LDIM + g * 16 + n] / a_scale[kk] : 0.f;
    } else {
      const int g = f - F_C;
      v = Cw[((g >> 1) * 32 + kk) * SDIM + (g & 1) * 16 + n];
    }
    wfg[idx] = (_Float16)v;
  }
  if (tid < LDIM) {
    // biasS[n] = b1[n] - sum_k (shift_k/scale_k) * W1[k,n]
    float accS = b1[tid];
    for (int k = 0; k < SDIM; ++k)
      accS -= (s_shift[k] / s_scale[k]) * W1[k * LDIM + tid];
    biasS[tid] = accS;
    // biasA[n] = - sum_k (ashift_k/ascale_k) * B[k,n]
    float accA = 0.f;
    for (int k = 0; k < ADIM; ++k)
      accA -= (a_shift[k] / a_scale[k]) * Bw[k * LDIM + tid];
    biasA[tid] = accA;
  }
}

// ---------------------------------------------------------------------------
// Main fused kernel: one 16-row tile per wave, 8 waves/block.
// ---------------------------------------------------------------------------
__global__ __launch_bounds__(256) void koopman_fused_kernel(
    const float* __restrict__ states, const float* __restrict__ actions,
    const float* __restrict__ b2, const float* __restrict__ b3,
    const _Float16* __restrict__ wfg, const float* __restrict__ biasS,
    const float* __restrict__ biasA,
    float* __restrict__ pred, float* __restrict__ latent) {
  __shared__ __attribute__((aligned(16))) _Float16 wf[NFRAG * 512];   // 36 KB
  __shared__ __attribute__((aligned(16))) _Float16 stage[WPB][16 * PITCH]; // 18 KB

  const int tid = threadIdx.x;

  // fast coalesced copy of prebuilt fragments: 2304 uint4, 9 per thread
  {
    const uint4* src = (const uint4*)wfg;
    uint4* dst = (uint4*)wf;
#pragma unroll
    for (int i = 0; i < (NFRAG * 512 * 2) / (16 * 256); ++i)
      dst[tid + i * 256] = src[tid + i * 256];
  }
  __syncthreads();

  const int wave = tid >> 5;
  const int lane = tid & 31;
  const int m    = lane & 15;           // A-frag row / C-frag column identity
  const int kb   = (lane >> 4) << 3;    // A-frag K sub-base (0 or 8)
  const int mr0  = (lane >> 4) << 3;    // C-frag row base (0 or 8)
  _Float16* st   = stage[wave];

  const int tile = blockIdx.x * WPB + wave;            // grid launched exactly
  const size_t r0 = (size_t)tile * 16;                 // global row base
  const int t0   = (int)(r0 & (SEQ - 1));              // timestep base (tile never crosses seq)
  const int bix  = (int)(r0 >> 10);                    // batch index

  // per-lane bias values (L2-resident)
  float b1v[4], b2v[4], b3v[4], bAv[4];
#pragma unroll
  for (int nt = 0; nt < 4; ++nt) {
    b1v[nt] = biasS[nt * 16 + m];
    b2v[nt] = b2[nt * 16 + m];
    b3v[nt] = b3[nt * 16 + m];
    bAv[nt] = biasA[nt * 16 + m];
  }

  // ================= stage 1: x A-frag straight from global ==============
  v16h xf;
  {
    const float* sp = states + (r0 + (size_t)m) * SDIM;
#pragma unroll
    for (int i = 0; i < 8; ++i) {
      xf[i]     = (_Float16)sp[kb + i];
      xf[i + 8] = (_Float16)sp[kb + 16 + i];
    }
  }
  // h1 = relu(x @ W1' + b1') -> staging
#pragma unroll
  for (int nt = 0; nt < 4; ++nt) {
    v8f c = {};
    c = wmma_f16(xf, bfrag(wf, F_W1 + nt, lane), c);
#pragma unroll
    for (int j = 0; j < 8; ++j) {
      float v = c[j] + b1v[nt];
      st[(mr0 + j) * PITCH + nt * 16 + m] = (_Float16)(v > 0.f ? v : 0.f);
    }
  }
  __builtin_amdgcn_wave_barrier();

  // ================= stage 2: h2 = relu(h1 @ W2 + b2) ====================
  v16h a0 = afrag(st, m, kb, 0);
  v16h a1 = afrag(st, m, kb, 1);
  __builtin_amdgcn_wave_barrier();
#pragma unroll
  for (int nt = 0; nt < 4; ++nt) {
    v8f c = {};
    c = wmma_f16(a0, bfrag(wf, F_W2 + nt, lane), c);
    c = wmma_f16(a1, bfrag(wf, F_W2 + 4 + nt, lane), c);
#pragma unroll
    for (int j = 0; j < 8; ++j) {
      float v = c[j] + b2v[nt];
      st[(mr0 + j) * PITCH + nt * 16 + m] = (_Float16)(v > 0.f ? v : 0.f);
    }
  }
  __builtin_amdgcn_wave_barrier();

  // ================= stage 3: z = h2 @ W3 + b3  (latent output) ==========
  a0 = afrag(st, m, kb, 0);
  a1 = afrag(st, m, kb, 1);
  __builtin_amdgcn_wave_barrier();
#pragma unroll
  for (int nt = 0; nt < 4; ++nt) {
    v8f c = {};
    c = wmma_f16(a0, bfrag(wf, F_W3 + nt, lane), c);
    c = wmma_f16(a1, bfrag(wf, F_W3 + 4 + nt, lane), c);
#pragma unroll
    for (int j = 0; j < 8; ++j) {
      const float v = c[j] + b3v[nt];
      latent[(r0 + (size_t)(mr0 + j)) * LDIM + nt * 16 + m] = v;   // f32 out
      st[(mr0 + j) * PITCH + nt * 16 + m] = (_Float16)v;           // f16 stage
    }
  }
  __builtin_amdgcn_wave_barrier();

  // ================= stage 4: nl = z @ A + act @ B' + biasA ===============
  a0 = afrag(st, m, kb, 0);
  a1 = afrag(st, m, kb, 1);
  __builtin_amdgcn_wave_barrier();
  // action A-frag straight from global: only lanes 0-15, elements 0-7 map K<8
  v16h af = {};
  if (lane < 16) {
    const int t = t0 + lane;
    if (t < SEQ - 1) {
      const float* ap = actions + ((size_t)bix * (SEQ - 1) + t) * ADIM;
#pragma unroll
      for (int i = 0; i < ADIM; ++i) af[i] = (_Float16)ap[i];
    }
  }
#pragma unroll
  for (int nt = 0; nt < 4; ++nt) {
    v8f c = {};
    c = wmma_f16(a0, bfrag(wf, F_A + nt, lane), c);
    c = wmma_f16(a1, bfrag(wf, F_A + 4 + nt, lane), c);
    c = wmma_f16(af, bfrag(wf, F_B + nt, lane), c);
#pragma unroll
    for (int j = 0; j < 8; ++j)
      st[(mr0 + j) * PITCH + nt * 16 + m] = (_Float16)(c[j] + bAv[nt]);
  }
  __builtin_amdgcn_wave_barrier();

  // ================= stage 5: predictions = nl @ C ========================
  a0 = afrag(st, m, kb, 0);
  a1 = afrag(st, m, kb, 1);
#pragma unroll
  for (int nt = 0; nt < 2; ++nt) {
    v8f c = {};
    c = wmma_f16(a0, bfrag(wf, F_C + nt, lane), c);
    c = wmma_f16(a1, bfrag(wf, F_C + 2 + nt, lane), c);
#pragma unroll
    for (int j = 0; j < 8; ++j) {
      const int t = t0 + mr0 + j;
      if (t < SEQ - 1)                                  // last timestep has no prediction
        pred[((size_t)bix * (SEQ - 1) + t) * SDIM + nt * 16 + m] = c[j];
    }
  }
}

extern "C" void kernel_launch(void* const* d_in, const int* in_sizes, int n_in,
                              void* d_out, int out_size, void* d_ws, size_t ws_size,
                              hipStream_t stream) {
  (void)in_sizes; (void)n_in; (void)out_size; (void)ws_size;
  const float* states  = (const float*)d_in[0];
  const float* actions = (const float*)d_in[1];
  const float* W1      = (const float*)d_in[2];
  const float* b1      = (const float*)d_in[3];
  const float* W2      = (const float*)d_in[4];
  const float* b2      = (const float*)d_in[5];
  const float* W3      = (const float*)d_in[6];
  const float* b3      = (const float*)d_in[7];
  const float* Aw      = (const float*)d_in[8];
  const float* Bw      = (const float*)d_in[9];
  const float* Cw      = (const float*)d_in[10];
  const float* s_shift = (const float*)d_in[11];
  const float* s_scale = (const float*)d_in[12];
  const float* a_shift = (const float*)d_in[13];
  const float* a_scale = (const float*)d_in[14];

  float* pred   = (float*)d_out;                                   // [256,1023,32]
  float* latent = pred + (size_t)BATCH * (SEQ - 1) * SDIM;         // [256,1024,64]

  _Float16* wfg  = (_Float16*)((char*)d_ws + WS_WF);
  float*    bS   = (float*)((char*)d_ws + WS_BIASS);
  float*    bA   = (float*)((char*)d_ws + WS_BIASA);

  koopman_prep_kernel<<<1, 256, 0, stream>>>(
      W1, b1, W2, W3, Aw, Bw, Cw, s_shift, s_scale, a_shift, a_scale,
      wfg, bS, bA);

  const int blocks = TILES / WPB;   // 16384 tiles / 8 waves = 2048 blocks, exact
  koopman_fused_kernel<<<blocks, 256, 0, stream>>>(
      states, actions, b2, b3, wfg, bS, bA, pred, latent);
}
